// SelfAttention_10668698764057
// MI455X (gfx1250) — compile-verified, matrix operand order checked by reading, not run on hardware
//
#include <hip/hip_runtime.h>
#include <hip/hip_bf16.h>

// ---------------------------------------------------------------------------
// MI455X (gfx1250, wave32) implementation.
// Pipeline: LN1 -> QKV gemm (WMMA f16) -> attn scores+softmax (VALU, tiny)
//           -> apply-attn (WMMA) -> proj (WMMA) -> LN2 -> fused FFN (WMMA)
// All GEMMs: f16 inputs, f32 accumulation via v_wmma_f32_16x16x32_f16.
// Inner loops use explicit two-stage ping-pong buffering (k unrolled by 64)
// so fragment loads overlap WMMAs with NO register-rotation copies.
// ---------------------------------------------------------------------------

typedef _Float16 f16;
typedef __attribute__((ext_vector_type(16))) _Float16 v16h;
typedef __attribute__((ext_vector_type(8)))  _Float16 v8h;
typedef __attribute__((ext_vector_type(8)))  float    v8f;

#define Bq   8
#define Cq   256
#define Nq   4096      // H*W
#define Mtot 32768     // B*N
#define FFNq 2048

__device__ __forceinline__ v8f zero8() { v8f z = {}; return z; }

__device__ __forceinline__ v16h combine16(v8h lo, v8h hi) {
  v16h r;
#pragma unroll
  for (int i = 0; i < 8; ++i) { r[i] = lo[i]; r[8 + i] = hi[i]; }
  return r;
}

// Fragment loader for 16-bit WMMA A/B operands.
// A layout (16x32, MxK): lane holds row (lane&15); lanes 0-15 K in {0..7,16..23},
// lanes 16-31 K in {8..15,24..31}.  B (32x16, KxN) from a row-major (Nout x K)
// weight matrix has the identical access pattern with "row" = output column.
__device__ __forceinline__ v16h load_frag(const f16* base, int ld, int lane) {
  const f16* p = base + (long)(lane & 15) * ld + ((lane & 16) >> 1); // +8 elems for hi half
  v8h lo = *(const v8h*)(p);
  v8h hi = *(const v8h*)(p + 16);
  return combine16(lo, hi);
}

__device__ __forceinline__ v8f wmma_f16(v16h a, v16h b, v8f c) {
  return __builtin_amdgcn_wmma_f32_16x16x32_f16(false, a, false, b, (short)0, c,
                                                false, false);
}

// ---------------------------------------------------------------------------
// f32 -> f16 weight conversion
// ---------------------------------------------------------------------------
__global__ void cvt_f16_kernel(const float* __restrict__ in, f16* __restrict__ out, int n) {
  int i = blockIdx.x * 256 + threadIdx.x;
  if (i < n) out[i] = (f16)in[i];
}

// ---------------------------------------------------------------------------
// LN1: x (B,C,N) f32 -> xln (B*N, C) f16   (channel LayerNorm per pixel)
// Block = 256 threads, tile = 64 pixels; thread (px, grp) covers 64 channels.
// ---------------------------------------------------------------------------
__global__ void __launch_bounds__(256) ln1_kernel(
    const float* __restrict__ x, const float* __restrict__ g,
    const float* __restrict__ be, f16* __restrict__ xln) {
  __shared__ float ps[4][64], ps2[4][64], muv[64], rsv[64];
  int t = threadIdx.x;
  int px = t & 63, grp = t >> 6;
  long pg = (long)blockIdx.x * 64 + px;     // global pixel
  int b = (int)(pg >> 12);
  int n = (int)(pg & 4095);
  const float* xb = x + (long)b * Cq * Nq + n;
  float s = 0.f, s2 = 0.f;
  for (int c = grp * 64; c < grp * 64 + 64; ++c) {
    float v = xb[(long)c * Nq];
    s += v; s2 += v * v;
  }
  ps[grp][px] = s; ps2[grp][px] = s2;
  __syncthreads();
  if (t < 64) {
    float S = 0.f, S2 = 0.f;
#pragma unroll
    for (int q = 0; q < 4; ++q) { S += ps[q][t]; S2 += ps2[q][t]; }
    float mu = S * (1.f / 256.f);
    float var = S2 * (1.f / 256.f) - mu * mu;
    muv[t] = mu;
    rsv[t] = rsqrtf(var + 1e-5f);
  }
  __syncthreads();
  float mu = muv[px], rs = rsv[px];
  f16* ob = xln + pg * Cq + grp * 64;
  for (int cc = 0; cc < 64; cc += 8) {
    v8h o;
#pragma unroll
    for (int jj = 0; jj < 8; ++jj) {
      int c = grp * 64 + cc + jj;
      float v = xb[(long)c * Nq];
      o[jj] = (f16)((v - mu) * rs * g[c] + be[c]);
    }
    *(v8h*)(ob + cc) = o;
  }
}

// ---------------------------------------------------------------------------
// LN2: in (B*N, C) f16 -> out (B*N, C) f16.  One wave per pixel.
// ---------------------------------------------------------------------------
__global__ void __launch_bounds__(256) ln2_kernel(
    const f16* __restrict__ in, const float* __restrict__ g,
    const float* __restrict__ be, f16* __restrict__ out) {
  int lane = threadIdx.x & 31, wave = threadIdx.x >> 5;
  long pix = (long)blockIdx.x * 8 + wave;
  const f16* ib = in + pix * Cq + lane * 8;
  v8h v = *(const v8h*)ib;
  float s = 0.f, s2 = 0.f;
#pragma unroll
  for (int j = 0; j < 8; ++j) { float f = (float)v[j]; s += f; s2 += f * f; }
#pragma unroll
  for (int off = 16; off >= 1; off >>= 1) {
    s  += __shfl_xor(s,  off, 32);
    s2 += __shfl_xor(s2, off, 32);
  }
  float mu = s * (1.f / 256.f);
  float var = s2 * (1.f / 256.f) - mu * mu;
  float rs = rsqrtf(var + 1e-5f);
  v8h o;
#pragma unroll
  for (int j = 0; j < 8; ++j) {
    int c = lane * 8 + j;
    o[j] = (f16)(((float)v[j] - mu) * rs * g[c] + be[c]);
  }
  *(v8h*)(out + pix * Cq + lane * 8) = o;
}

// ---------------------------------------------------------------------------
// Generic WMMA GEMM: C[M x Nout] = A[M x K] * W^T + bias, W is (Nout x K) f16
// row-major.  Block = 256 threads (8 waves); wave tile = 16(M) x 64(N).
// grid = (M/128, Nout/64, Z).  Per-operand z offset = (z/zmod)*sOut+(z%zmod)*sIn.
// K must be a multiple of 64.  Two-stage ping-pong: stage-0/1 fragments live
// in distinct registers, refilled directly by loads (no rotation copies).
// ---------------------------------------------------------------------------
__global__ void __launch_bounds__(256) gemm_wmma(
    const f16* __restrict__ A, int lda, long sAo, long sAi,
    const f16* __restrict__ W, int ldw, long sWo, long sWi,
    const float* __restrict__ bias, int relu,
    f16* __restrict__ C, int ldc, long sCo, long sCi,
    int zmod, int K) {
  int lane = threadIdx.x & 31, wave = threadIdx.x >> 5;
  int z = blockIdx.z;
  long za = (long)(z / zmod) * sAo + (long)(z % zmod) * sAi;
  long zw = (long)(z / zmod) * sWo + (long)(z % zmod) * sWi;
  long zc = (long)(z / zmod) * sCo + (long)(z % zmod) * sCi;
  long m0 = (long)blockIdx.x * 128 + wave * 16;
  const f16* Ab = A + za + m0 * lda;
  const f16* Wb = W + zw + (long)blockIdx.y * 64 * ldw;

  v8f acc[4];
#pragma unroll
  for (int i = 0; i < 4; ++i) acc[i] = zero8();

  // stage 0 prologue (k = 0)
  v16h af0 = load_frag(Ab, lda, lane);
  v16h bf0[4];
#pragma unroll
  for (int nt = 0; nt < 4; ++nt)
    bf0[nt] = load_frag(Wb + (long)nt * 16 * ldw, ldw, lane);

  for (int kk = 0; kk < K; kk += 64) {
    // stage 1 loads (kk+32) -- always valid since K % 64 == 0
    __builtin_prefetch(Ab + (long)(lane & 15) * lda + kk + 64, 0, 3);
    v16h af1 = load_frag(Ab + kk + 32, lda, lane);
    v16h bf1[4];
#pragma unroll
    for (int nt = 0; nt < 4; ++nt)
      bf1[nt] = load_frag(Wb + (long)nt * 16 * ldw + kk + 32, ldw, lane);

    // stage 0 WMMAs
#pragma unroll
    for (int nt = 0; nt < 4; ++nt)
      acc[nt] = wmma_f16(af0, bf0[nt], acc[nt]);

    // stage 0 refill for next iteration (kk+64)
    if (kk + 64 < K) {
      af0 = load_frag(Ab + kk + 64, lda, lane);
#pragma unroll
      for (int nt = 0; nt < 4; ++nt)
        bf0[nt] = load_frag(Wb + (long)nt * 16 * ldw + kk + 64, ldw, lane);
    }

    // stage 1 WMMAs
#pragma unroll
    for (int nt = 0; nt < 4; ++nt)
      acc[nt] = wmma_f16(af1, bf1[nt], acc[nt]);
  }

  int col = lane & 15, half = lane >> 4;
  f16* Cb = C + zc + (m0 + half * 8) * ldc + (long)blockIdx.y * 64;
#pragma unroll
  for (int nt = 0; nt < 4; ++nt) {
    float bv = bias ? bias[blockIdx.y * 64 + nt * 16 + col] : 0.f;
#pragma unroll
    for (int r = 0; r < 8; ++r) {
      float v = acc[nt][r] + bv;
      if (relu) v = fmaxf(v, 0.f);
      Cb[(long)r * ldc + nt * 16 + col] = (f16)v;
    }
  }
}

// ---------------------------------------------------------------------------
// Attention scores + softmax per (b, head): attn[i][j] =
// softmax_j( 0.125 * sum_n q[b,n,h*64+i] * k[b,n,h*64+j] ).  Grid = 32.
// ---------------------------------------------------------------------------
__global__ void __launch_bounds__(256) attn_scores(
    const f16* __restrict__ qkv, f16* __restrict__ attn) {
  __shared__ f16 qs[64 * 128];
  __shared__ f16 ks[64 * 128];
  __shared__ float sc[64 * 64];
  int z = blockIdx.x;
  int b = z >> 2, h = z & 3;
  int t = threadIdx.x;
  int i = t >> 2, j0 = (t & 3) * 16;
  const f16* qb = qkv + (long)b * Nq * 768 + h * 64;
  const f16* kb = qb + 256;

  float acc[16];
#pragma unroll
  for (int jj = 0; jj < 16; ++jj) acc[jj] = 0.f;

  for (int n0 = 0; n0 < Nq; n0 += 128) {
    __syncthreads();
    for (int idx = t; idx < 128 * 8; idx += 256) {
      int p = idx >> 3, ch = idx & 7;
      v8h qv = *(const v8h*)(qb + (long)(n0 + p) * 768 + ch * 8);
      v8h kv = *(const v8h*)(kb + (long)(n0 + p) * 768 + ch * 8);
#pragma unroll
      for (int jj = 0; jj < 8; ++jj) {
        qs[(ch * 8 + jj) * 128 + p] = qv[jj];
        ks[(ch * 8 + jj) * 128 + p] = kv[jj];
      }
    }
    __syncthreads();
    for (int n = 0; n < 128; ++n) {
      float qv = (float)qs[i * 128 + n];
#pragma unroll
      for (int jj = 0; jj < 16; ++jj)
        acc[jj] += qv * (float)ks[(j0 + jj) * 128 + n];
    }
  }
#pragma unroll
  for (int jj = 0; jj < 16; ++jj) sc[i * 64 + j0 + jj] = acc[jj] * 0.125f;
  __syncthreads();

  if (t < 64) {
    float mx = -1e30f;
    for (int j = 0; j < 64; ++j) mx = fmaxf(mx, sc[t * 64 + j]);
    float s = 0.f;
    for (int j = 0; j < 64; ++j) s += __expf(sc[t * 64 + j] - mx);
    float inv = 1.f / s;
    f16* ab = attn + ((long)z * 64 + t) * 64;
    for (int j = 0; j < 64; ++j) ab[j] = (f16)(__expf(sc[t * 64 + j] - mx) * inv);
  }
}

// ---------------------------------------------------------------------------
// Fused FFN + residual + transpose:
//   out[b,c,n] = (relu(y @ W1^T + b1) @ W2^T + b2)[n,c] + x[b,c,n]
// Wave tile = 16 pixels; FFN chunked by 32; hidden tile bounced through LDS
// to re-layout WMMA D -> A fragments.  Grid = 256 blocks of 256 threads.
// Y-tile A-fragments are invariant across FFN chunks -> hoisted into yf[8].
// ---------------------------------------------------------------------------
__global__ void __launch_bounds__(256) ffn_fused(
    const f16* __restrict__ Y, const f16* __restrict__ W1,
    const float* __restrict__ b1, const f16* __restrict__ W2,
    const float* __restrict__ b2, const float* __restrict__ x,
    float* __restrict__ out) {
  __shared__ __align__(16) f16 hst[8][16 * 32];
  int lane = threadIdx.x & 31, wave = threadIdx.x >> 5;
  int m0 = blockIdx.x * 128 + wave * 16;
  int b = m0 >> 12;
  int pix0 = m0 & 4095;
  const f16* Yb = Y + (long)m0 * Cq;
  int col = lane & 15, half = lane >> 4;

  v8f acc[16];
#pragma unroll
  for (int i = 0; i < 16; ++i) acc[i] = zero8();

  // Hoist the 16x256 Y tile as 8 A-fragments (reused by every FFN chunk).
  v16h yf[8];
#pragma unroll
  for (int s = 0; s < 8; ++s) yf[s] = load_frag(Yb + s * 32, Cq, lane);

  for (int f0 = 0; f0 < FFNq; f0 += 32) {
    // ---- FFN1: h[m, f0..f0+31] = relu(y_tile @ W1^T + b1) ----
    v8f h0 = zero8(), h1 = zero8();
    v16h ba = load_frag(W1 + (long)f0 * Cq, Cq, lane);
    v16h bb = load_frag(W1 + (long)(f0 + 16) * Cq, Cq, lane);
#pragma unroll
    for (int s = 0; s < 8; ++s) {   // fully unrolled: rotation renames, no movs
      v16h ba_n = ba, bb_n = bb;
      if (s + 1 < 8) {
        ba_n = load_frag(W1 + (long)f0 * Cq + (s + 1) * 32, Cq, lane);
        bb_n = load_frag(W1 + (long)(f0 + 16) * Cq + (s + 1) * 32, Cq, lane);
      }
      h0 = wmma_f16(yf[s], ba, h0);
      h1 = wmma_f16(yf[s], bb, h1);
      ba = ba_n; bb = bb_n;
    }
    float bb0 = b1[f0 + col], bb1 = b1[f0 + 16 + col];
#pragma unroll
    for (int r = 0; r < 8; ++r) {
      int m = r + half * 8;
      hst[wave][m * 32 + col]      = (f16)fmaxf(h0[r] + bb0, 0.f);
      hst[wave][m * 32 + 16 + col] = (f16)fmaxf(h1[r] + bb1, 0.f);
    }
    asm volatile("s_wait_dscnt 0" ::: "memory");  // same-wave LDS RAW

    // Re-read as A fragment (16x32 f16).
    const f16* p = &hst[wave][(lane & 15) * 32 + ((lane & 16) >> 1)];
    v16h a2 = combine16(*(const v8h*)p, *(const v8h*)(p + 16));

    // ---- FFN2: acc += h_chunk @ W2^T ----  (pipelined over nt, unrolled)
    v16h bf2 = load_frag(W2 + f0, FFNq, lane);
#pragma unroll
    for (int nt = 0; nt < 16; ++nt) {
      v16h bf2_n = bf2;
      if (nt + 1 < 16)
        bf2_n = load_frag(W2 + (long)((nt + 1) * 16) * FFNq + f0, FFNq, lane);
      acc[nt] = wmma_f16(a2, bf2, acc[nt]);
      bf2 = bf2_n;
    }
  }

  // Epilogue: + b2 + residual x, transposed store to (B,C,H,W).
#pragma unroll
  for (int nt = 0; nt < 16; ++nt) {
    int c = nt * 16 + col;
    float bv = b2[c];
    long base = ((long)(b * Cq + c)) * Nq + pix0 + half * 8;
    float4 r0 = *(const float4*)(x + base);
    float4 r1 = *(const float4*)(x + base + 4);
    float4 o0, o1;
    o0.x = acc[nt][0] + bv + r0.x;
    o0.y = acc[nt][1] + bv + r0.y;
    o0.z = acc[nt][2] + bv + r0.z;
    o0.w = acc[nt][3] + bv + r0.w;
    o1.x = acc[nt][4] + bv + r1.x;
    o1.y = acc[nt][5] + bv + r1.y;
    o1.z = acc[nt][6] + bv + r1.z;
    o1.w = acc[nt][7] + bv + r1.w;
    *(float4*)(out + base)     = o0;
    *(float4*)(out + base + 4) = o1;
  }
}

// ---------------------------------------------------------------------------
// Host launcher
// ---------------------------------------------------------------------------
extern "C" void kernel_launch(void* const* d_in, const int* in_sizes, int n_in,
                              void* d_out, int out_size, void* d_ws, size_t ws_size,
                              hipStream_t stream) {
  (void)in_sizes; (void)n_in; (void)out_size; (void)ws_size;
  const float* x      = (const float*)d_in[0];
  const float* w_qkv  = (const float*)d_in[1];
  const float* b_qkv  = (const float*)d_in[2];
  const float* w_proj = (const float*)d_in[3];
  const float* b_proj = (const float*)d_in[4];
  const float* w_ffn1 = (const float*)d_in[5];
  const float* b_ffn1 = (const float*)d_in[6];
  const float* w_ffn2 = (const float*)d_in[7];
  const float* b_ffn2 = (const float*)d_in[8];
  const float* g1     = (const float*)d_in[9];
  const float* be1    = (const float*)d_in[10];
  const float* g2     = (const float*)d_in[11];
  const float* be2    = (const float*)d_in[12];
  float* out = (float*)d_out;

  char* ws = (char*)d_ws;
  f16* wq16     = (f16*)(ws + 0);          //   768*256 f16
  f16* wp16     = (f16*)(ws + 393216);     //   256*256
  f16* w116     = (f16*)(ws + 524288);     //  2048*256
  f16* w216     = (f16*)(ws + 1572864);    //  256*2048
  f16* xln      = (f16*)(ws + 2621440);    //  32768*256  (reused as proj16)
  f16* qkv      = (f16*)(ws + 19398656);   //  32768*768  (reused as y)
  f16* attn16   = (f16*)(ws + 69730304);   //  32*64*64
  f16* attn_out = (f16*)(ws + 69992448);   //  32768*256  -> 86769664 total bytes

  f16* proj16 = xln;   // xln dead after QKV gemm
  f16* y      = qkv;   // qkv dead after apply-attn

  // 1) weights -> f16
  cvt_f16_kernel<<<768,  256, 0, stream>>>(w_qkv,  wq16, 768 * 256);
  cvt_f16_kernel<<<256,  256, 0, stream>>>(w_proj, wp16, 256 * 256);
  cvt_f16_kernel<<<2048, 256, 0, stream>>>(w_ffn1, w116, 2048 * 256);
  cvt_f16_kernel<<<2048, 256, 0, stream>>>(w_ffn2, w216, 256 * 2048);

  // 2) LN1 (transpose (B,C,N) -> (B*N,C) f16)
  ln1_kernel<<<Mtot / 64, 256, 0, stream>>>(x, g1, be1, xln);

  // 3) QKV gemm: (32768x256) x (768x256)^T
  gemm_wmma<<<dim3(Mtot / 128, 768 / 64, 1), 256, 0, stream>>>(
      xln, 256, 0, 0, wq16, 256, 0, 0, b_qkv, 0, qkv, 768, 0, 0, 1, 256);

  // 4) attention scores + softmax, per (b, head)
  attn_scores<<<32, 256, 0, stream>>>(qkv, attn16);

  // 5) apply attention: per (b,h) gemm, A = v slice, W = attn (64x64)
  gemm_wmma<<<dim3(Nq / 128, 1, 32), 256, 0, stream>>>(
      qkv + 512, 768, (long)Nq * 768, 64,
      attn16, 64, 4 * 64 * 64, 64 * 64,
      nullptr, 0,
      attn_out, 256, (long)Nq * 256, 64,
      4, 64);

  // 6) proj gemm: (32768x256) x (256x256)^T
  gemm_wmma<<<dim3(Mtot / 128, 256 / 64, 1), 256, 0, stream>>>(
      attn_out, 256, 0, 0, wp16, 256, 0, 0, b_proj, 0, proj16, 256, 0, 0, 1, 256);

  // 7) LN2
  ln2_kernel<<<Mtot / 8, 256, 0, stream>>>(proj16, g2, be2, y);

  // 8) fused FFN + bias + residual + transpose store
  ffn_fused<<<Mtot / 128, 256, 0, stream>>>(y, w116, b_ffn1, w216, b_ffn2, x, out);
}
